// AttentionWithDropout_34626026340718
// MI455X (gfx1250) — matrix operand order repaired and checked by reading, not compiled, not run on hardware
//
#include <hip/hip_runtime.h>
#include <hip/hip_bf16.h>

typedef _Float16 half_t;
typedef __attribute__((ext_vector_type(16))) _Float16 v16h;
typedef __attribute__((ext_vector_type(8)))  _Float16 v8h;
typedef __attribute__((ext_vector_type(8)))  float    v8f;
typedef __attribute__((ext_vector_type(4)))  unsigned int v4u;
typedef __attribute__((ext_vector_type(8)))  int v8i_t;
typedef __attribute__((ext_vector_type(4)))  int v4i_t;

#define E_DIM  1024
#define H_NUM  16
#define D_HEAD 64
#define L_SEQ  2048
#define B_SZ   2
#define M_TOT  (B_SZ * L_SEQ)   // 4096

union AFrag {
  v16h v;
  v8h  h[2];
};

__device__ __forceinline__ v8f wmma_f16(const AFrag& a, const AFrag& b, v8f c) {
  // D = A(16x32 f16) * B(32x16 f16) + C(16x16 f32)
  return __builtin_amdgcn_wmma_f32_16x16x32_f16(false, a.v, false, b.v,
                                                (short)0, c, false, false);
}

__device__ __forceinline__ void cvt16_f32_to_f16(const float* __restrict__ s,
                                                 half_t* __restrict__ d) {
  const float4* s4 = (const float4*)s;
  float4 a = s4[0], b = s4[1], c = s4[2], e = s4[3];
  v8h lo, hi;
  lo[0] = (half_t)a.x; lo[1] = (half_t)a.y; lo[2] = (half_t)a.z; lo[3] = (half_t)a.w;
  lo[4] = (half_t)b.x; lo[5] = (half_t)b.y; lo[6] = (half_t)b.z; lo[7] = (half_t)b.w;
  hi[0] = (half_t)c.x; hi[1] = (half_t)c.y; hi[2] = (half_t)c.z; hi[3] = (half_t)c.w;
  hi[4] = (half_t)e.x; hi[5] = (half_t)e.y; hi[6] = (half_t)e.z; hi[7] = (half_t)e.w;
  *(v8h*)d       = lo;
  *(v8h*)(d + 8) = hi;
}

// ---------------------------------------------------------------------------
// Tensor Data Mover: 2D f16 tile, global -> LDS, with LDS row padding.
// D# layout per CDNA5 ISA 8.3/8.4:
//   group0: count=1 | lds_addr | global_addr[56:0] | type=2
//   group1: data_size=2B, pad_enable, pad_interval, pad_amount,
//           tensor_dim0/1 (large, tiles always in-bounds),
//           tile_dim0 (row elems), tile_dim1 (rows), tensor_dim0_stride.
// pad_interval code n -> pad every (1<<n)*8 bytes; pad_amount code n -> (n+1) DWORDs.
// ---------------------------------------------------------------------------
__device__ __forceinline__ void tdm_load_tile_2d_f16(
    const half_t* gsrc, unsigned int lds_byte_off,
    unsigned int tile_elems_x, unsigned int tile_rows,
    unsigned long long row_stride_elems,
    unsigned int pad_interval_code, unsigned int pad_amount_code) {
  const unsigned long long ga = (unsigned long long)(uintptr_t)gsrc;
  v4u g0;
  g0[0] = 1u;                                                  // count=1 (user D#)
  g0[1] = lds_byte_off;                                        // lds_addr
  g0[2] = (unsigned int)ga;                                    // global_addr[31:0]
  g0[3] = (unsigned int)((ga >> 32) & 0x01ffffffu) | (2u << 30);  // [56:32] | type=2

  const unsigned int dim0 = 1u << 20;   // tensor dims: large, tiles in-bounds
  const unsigned int dim1 = 1u << 20;
  const unsigned int w0 = (1u << 16)                    // data_size = 2 bytes
                        | (1u << 20)                    // pad_enable
                        | (pad_interval_code << 22)
                        | (pad_amount_code << 25);
  v8i_t g1;
  g1[0] = (int)w0;
  g1[1] = (int)((dim0 & 0xffffu) << 16);                               // dim0 lo16
  g1[2] = (int)(((dim0 >> 16) & 0xffffu) | ((dim1 & 0xffffu) << 16));  // dim0 hi / dim1 lo
  g1[3] = (int)(((dim1 >> 16) & 0xffffu) | ((tile_elems_x & 0xffffu) << 16));
  g1[4] = (int)(tile_rows & 0xffffu);                                  // tile_dim1; dim2=0
  g1[5] = (int)(unsigned int)(row_stride_elems & 0xffffffffu);         // stride0[31:0]
  g1[6] = (int)(unsigned int)((row_stride_elems >> 32) & 0xffffu);     // stride0[47:32]
  g1[7] = 0;
  const v4i_t gz = {0, 0, 0, 0};
#if defined(__clang_major__) && __clang_major__ >= 23
  const v8i_t gx = {0, 0, 0, 0, 0, 0, 0, 0};
  __builtin_amdgcn_tensor_load_to_lds(g0, g1, gz, gz, gx, 0);
#else
  __builtin_amdgcn_tensor_load_to_lds(g0, g1, gz, gz, 0);
#endif
}

// ---------------------------------------------------------------------------
// Tiled GEMM: out[m][n] = sum_k A[m][k] * W[n][k] + bias[n]
//   MODE 0: out = f16, layout [B, H, L, D]      (Q, K projections)
//   MODE 1: out = f16, layout [B, H, D, L]      (V projection, transposed)
//   MODE 2: out = f32, layout [M, N] row-major  (final output projection)
// A_F16: A supplied as f16 (attention output) instead of f32.
// Block: 128 threads = 4 waves; 64x64 block tile; wave = 32x32 (2x2 WMMA).
// ---------------------------------------------------------------------------
template <int MODE, bool A_F16>
__global__ __launch_bounds__(128)
void gemm_wmma_kernel(const float* __restrict__ Af, const half_t* __restrict__ Ah,
                      const float* __restrict__ Wf, const float* __restrict__ bias,
                      half_t* __restrict__ out16, float* __restrict__ out32,
                      int M, int N, int K) {
  __shared__ __attribute__((aligned(16))) half_t lsA[64][40];
  __shared__ __attribute__((aligned(16))) half_t lsW[64][40];

  const int t    = threadIdx.x;
  const int lane = t & 31;
  const int wave = t >> 5;
  const int wr   = wave >> 1;     // wave row within 2x2 wave grid
  const int wc   = wave & 1;      // wave col
  const int ln   = lane & 15;     // lane-in-16
  const int lh   = lane >> 4;     // half-wave select
  const int m0   = blockIdx.y * 64;
  const int n0   = blockIdx.x * 64;

  const v8f vzero = {0.f, 0.f, 0.f, 0.f, 0.f, 0.f, 0.f, 0.f};
  v8f acc[2][2];
#pragma unroll
  for (int i = 0; i < 2; ++i)
#pragma unroll
    for (int j = 0; j < 2; ++j) acc[i][j] = vzero;

  const int row = t >> 1;   // 0..63: tile row staged by this thread
  const int seg = t & 1;    // 0..1 : 16-column segment

  for (int kk = 0; kk < K; kk += 32) {
    // ---- stage A tile (64 x 32) into LDS, converting to f16 if needed ----
    if (A_F16) {
      const v8h* src = (const v8h*)(Ah + (size_t)(m0 + row) * K + kk + seg * 16);
      *(v8h*)&lsA[row][seg * 16]     = src[0];
      *(v8h*)&lsA[row][seg * 16 + 8] = src[1];
    } else {
      cvt16_f32_to_f16(Af + (size_t)(m0 + row) * K + kk + seg * 16,
                       &lsA[row][seg * 16]);
    }
    // ---- stage W tile (64 x 32, W stored N x K row-major) ----
    cvt16_f32_to_f16(Wf + (size_t)(n0 + row) * K + kk + seg * 16,
                     &lsW[row][seg * 16]);

    if (kk + 32 < K) {  // prefetch next K-step (global_prefetch_b8 path)
      __builtin_prefetch(Wf + (size_t)(n0 + row) * K + kk + 32 + seg * 16, 0, 1);
      if (A_F16)
        __builtin_prefetch(Ah + (size_t)(m0 + row) * K + kk + 32 + seg * 16, 0, 1);
      else
        __builtin_prefetch(Af + (size_t)(m0 + row) * K + kk + 32 + seg * 16, 0, 1);
    }
    __syncthreads();

    // ---- build fragments and issue 4 WMMAs ----
    AFrag a[2], b[2];
#pragma unroll
    for (int i = 0; i < 2; ++i) {
      const int r = wr * 32 + i * 16 + ln;      // A row (M) in lane
      a[i].h[0] = *(const v8h*)&lsA[r][8 * lh];         // K =  8*lh .. +7
      a[i].h[1] = *(const v8h*)&lsA[r][16 + 8 * lh];    // K = 16+8*lh .. +7
    }
#pragma unroll
    for (int j = 0; j < 2; ++j) {
      const int c = wc * 32 + j * 16 + ln;      // B col (N) in lane
      b[j].h[0] = *(const v8h*)&lsW[c][8 * lh];         // K =  8*lh .. +7
      b[j].h[1] = *(const v8h*)&lsW[c][16 + 8 * lh];    // K = 16+8*lh .. +7
    }
#pragma unroll
    for (int i = 0; i < 2; ++i)
#pragma unroll
      for (int j = 0; j < 2; ++j) acc[i][j] = wmma_f16(a[i], b[j], acc[i][j]);
    __syncthreads();
  }

  // ---- epilogue: bias + layout-specific store ----
#pragma unroll
  for (int i = 0; i < 2; ++i) {
#pragma unroll
    for (int j = 0; j < 2; ++j) {
      const int n  = n0 + wc * 32 + j * 16 + ln;
      const float bv = bias[n];
#pragma unroll
      for (int r = 0; r < 8; ++r) {
        const int m   = m0 + wr * 32 + i * 16 + r + 8 * lh;  // C-layout row
        const float v = acc[i][j][r] + bv;
        if (MODE == 2) {
          out32[(size_t)m * N + n] = v;
        } else {
          const int bb = m / L_SEQ, l = m % L_SEQ;
          const int hh = n / D_HEAD, d = n % D_HEAD;
          if (MODE == 0)
            out16[((size_t)(bb * H_NUM + hh) * L_SEQ + l) * D_HEAD + d] = (half_t)v;
          else
            out16[((size_t)(bb * H_NUM + hh) * D_HEAD + d) * L_SEQ + l] = (half_t)v;
        }
      }
    }
  }
}

// ---------------------------------------------------------------------------
// Flash attention: softmax(Q Kt / sqrt(E)) V per (b,h).
// Block: 128 threads = 4 waves; each wave owns 16 query rows.
// K/V tiles are DMA'd into double-buffered LDS by the Tensor Data Mover
// (wave 0 issues, s_wait_tensorcnt + barrier publishes), overlapping the DMA
// of tile i+1 with the 8 WMMAs + softmax of tile i. All 4 waves share the
// same LDS tile (4x less global traffic than per-wave fragment loads).
// q16,k16: [B,H,L,D] f16; vT16: [B,H,D,L] f16; out16: [B,L,E] f16.
// ---------------------------------------------------------------------------
__global__ __launch_bounds__(128)
void flash_attn_wmma_kernel(const half_t* __restrict__ q16,
                            const half_t* __restrict__ k16,
                            const half_t* __restrict__ vT16,
                            half_t* __restrict__ out16) {
  // TDM pads K rows (128B) with 16B -> stride 72 halfs; V rows (64B) with 16B -> 40 halfs.
  __shared__ __attribute__((aligned(16))) half_t lsK[2][32][72];
  __shared__ __attribute__((aligned(16))) half_t lsV[2][64][40];
  __shared__ __attribute__((aligned(16))) half_t lsP[4][16][40];  // wave-private

  const int t    = threadIdx.x;
  const int lane = t & 31;
  const int wave = t >> 5;
  const int ln   = lane & 15;
  const int lh   = lane >> 4;
  const int bh   = blockIdx.y;                    // b*H + h
  const int q0   = blockIdx.x * 64 + wave * 16;   // query-row tile origin

  const half_t* kh = k16 + (size_t)bh * L_SEQ * D_HEAD;   // [L, D] for this head
  const half_t* vh = vT16 + (size_t)bh * D_HEAD * L_SEQ;  // [D, L] for this head

  // Q fragments held in registers for the whole sweep (K-dim = D = 64 -> 2 frags)
  const half_t* qbase = q16 + ((size_t)bh * L_SEQ + q0 + ln) * D_HEAD;
  AFrag aq0, aq1;
  aq0.h[0] = *(const v8h*)(qbase + 8 * lh);
  aq0.h[1] = *(const v8h*)(qbase + 16 + 8 * lh);
  aq1.h[0] = *(const v8h*)(qbase + 32 + 8 * lh);
  aq1.h[1] = *(const v8h*)(qbase + 48 + 8 * lh);

  const v8f vzero = {0.f, 0.f, 0.f, 0.f, 0.f, 0.f, 0.f, 0.f};
  float mrow[8], lrow[8];
  v8f accO[4];
#pragma unroll
  for (int r = 0; r < 8; ++r) { mrow[r] = -1e30f; lrow[r] = 0.f; }
#pragma unroll
  for (int dt = 0; dt < 4; ++dt) accO[dt] = vzero;

  const float SCALE = 0.03125f;  // 1/sqrt(1024) — reference scales by sqrt(E)

  // ---- prologue: DMA first K/V tiles into buffer 0 ----
  if (wave == 0) {
    tdm_load_tile_2d_f16(kh, (unsigned int)(uintptr_t)&lsK[0][0][0],
                         /*row elems*/ 64, /*rows*/ 32, /*stride*/ 64,
                         /*pad: every 128B*/ 4, /*+4 DW*/ 3);
    tdm_load_tile_2d_f16(vh, (unsigned int)(uintptr_t)&lsV[0][0][0],
                         /*row elems*/ 32, /*rows*/ 64, /*stride*/ L_SEQ,
                         /*pad: every 64B*/ 3, /*+4 DW*/ 3);
  }

  for (int kk = 0; kk < L_SEQ; kk += 32) {
    const int p = (kk >> 5) & 1;
    if (wave == 0) __builtin_amdgcn_s_wait_tensorcnt(0);  // buffer p landed
    __syncthreads();  // publish buffer p; proves buffer p^1 reads (iter-1) done
    if (wave == 0 && kk + 32 < L_SEQ) {
      const int pn = p ^ 1;
      tdm_load_tile_2d_f16(kh + (size_t)(kk + 32) * D_HEAD,
                           (unsigned int)(uintptr_t)&lsK[pn][0][0],
                           64, 32, 64, 4, 3);
      tdm_load_tile_2d_f16(vh + (kk + 32),
                           (unsigned int)(uintptr_t)&lsV[pn][0][0],
                           32, 64, L_SEQ, 3, 3);
    }

    // ---- scores: two 16x16 tiles covering 32 key columns (from LDS) ----
    v8f s[2];
#pragma unroll
    for (int ct = 0; ct < 2; ++ct) {
      const int kr = ct * 16 + ln;          // key row within tile
      AFrag bk0, bk1;
      bk0.h[0] = *(const v8h*)&lsK[p][kr][8 * lh];        // d = 8*lh .. +7, 16+8*lh .. +7
      bk0.h[1] = *(const v8h*)&lsK[p][kr][16 + 8 * lh];
      bk1.h[0] = *(const v8h*)&lsK[p][kr][32 + 8 * lh];   // d = 32+8*lh .. +7, 48+8*lh .. +7
      bk1.h[1] = *(const v8h*)&lsK[p][kr][48 + 8 * lh];
      v8f z = vzero;
      z = wmma_f16(aq0, bk0, z);
      z = wmma_f16(aq1, bk1, z);
#pragma unroll
      for (int r = 0; r < 8; ++r) z[r] *= SCALE;
      s[ct] = z;
    }

    // ---- online softmax (row = element r + 8*lh, col = lane-in-16) ----
    float alpha[8];
#pragma unroll
    for (int r = 0; r < 8; ++r) {
      float tmax = fmaxf(s[0][r], s[1][r]);
#pragma unroll
      for (int msk = 1; msk < 16; msk <<= 1)
        tmax = fmaxf(tmax, __shfl_xor(tmax, msk, 32));
      const float mnew = fmaxf(mrow[r], tmax);
      alpha[r] = __expf(mrow[r] - mnew);
      const float p0 = __expf(s[0][r] - mnew);
      const float p1 = __expf(s[1][r] - mnew);
      s[0][r] = p0;
      s[1][r] = p1;
      float rs = p0 + p1;
#pragma unroll
      for (int msk = 1; msk < 16; msk <<= 1) rs += __shfl_xor(rs, msk, 32);
      lrow[r] = lrow[r] * alpha[r] + rs;
      mrow[r] = mnew;
    }

    // ---- transpose P: C-layout -> A-layout through wave-private LDS ----
#pragma unroll
    for (int ct = 0; ct < 2; ++ct)
#pragma unroll
      for (int r = 0; r < 8; ++r)
        lsP[wave][r + 8 * lh][ct * 16 + ln] = (half_t)s[ct][r];
    asm volatile("s_wait_dscnt 0x0" ::: "memory");   // wave-internal RAW on LDS
    AFrag ap;
    ap.h[0] = *(const v8h*)&lsP[wave][ln][8 * lh];
    ap.h[1] = *(const v8h*)&lsP[wave][ln][16 + 8 * lh];

    // ---- rescale O and accumulate P @ V (D = 64 -> 4 N-tiles, from LDS) ----
#pragma unroll
    for (int dt = 0; dt < 4; ++dt) {
#pragma unroll
      for (int r = 0; r < 8; ++r) accO[dt][r] *= alpha[r];
      AFrag bv;
      bv.h[0] = *(const v8h*)&lsV[p][dt * 16 + ln][8 * lh];
      bv.h[1] = *(const v8h*)&lsV[p][dt * 16 + ln][16 + 8 * lh];
      accO[dt] = wmma_f16(ap, bv, accO[dt]);
    }
  }

  // ---- finalize: divide by row sums, store [B, L, E] f16 ----
  const int b = bh / H_NUM, h = bh % H_NUM;
#pragma unroll
  for (int dt = 0; dt < 4; ++dt) {
#pragma unroll
    for (int r = 0; r < 8; ++r) {
      const float v   = accO[dt][r] / lrow[r];
      const int  qrow = q0 + r + 8 * lh;
      out16[((size_t)(b * L_SEQ + qrow)) * E_DIM + h * D_HEAD + dt * 16 + ln] =
          (half_t)v;
    }
  }
}

// ---------------------------------------------------------------------------
extern "C" void kernel_launch(void* const* d_in, const int* in_sizes, int n_in,
                              void* d_out, int out_size, void* d_ws, size_t ws_size,
                              hipStream_t stream) {
  const float* x_q = (const float*)d_in[0];
  const float* x_k = (const float*)d_in[1];
  const float* x_v = (const float*)d_in[2];
  const float* Wq  = (const float*)d_in[3];
  const float* bq  = (const float*)d_in[4];
  const float* Wk  = (const float*)d_in[5];
  const float* bk  = (const float*)d_in[6];
  const float* Wv  = (const float*)d_in[7];
  const float* bv  = (const float*)d_in[8];
  const float* Wo  = (const float*)d_in[9];
  const float* bo  = (const float*)d_in[10];

  char* ws = (char*)d_ws;
  const size_t seg = (size_t)M_TOT * E_DIM * sizeof(half_t);  // 8 MB each
  half_t* q16   = (half_t*)(ws);            // [B,H,L,D]
  half_t* k16   = (half_t*)(ws + seg);      // [B,H,L,D]
  half_t* vT16  = (half_t*)(ws + 2 * seg);  // [B,H,D,L]
  half_t* att16 = (half_t*)(ws + 3 * seg);  // [B,L,E]

  const dim3 gg(E_DIM / 64, M_TOT / 64);    // (16, 64)
  gemm_wmma_kernel<0, false><<<gg, 128, 0, stream>>>(
      x_q, nullptr, Wq, bq, q16, nullptr, M_TOT, E_DIM, E_DIM);
  gemm_wmma_kernel<0, false><<<gg, 128, 0, stream>>>(
      x_k, nullptr, Wk, bk, k16, nullptr, M_TOT, E_DIM, E_DIM);
  gemm_wmma_kernel<1, false><<<gg, 128, 0, stream>>>(
      x_v, nullptr, Wv, bv, vT16, nullptr, M_TOT, E_DIM, E_DIM);

  flash_attn_wmma_kernel<<<dim3(L_SEQ / 64, B_SZ * H_NUM), 128, 0, stream>>>(
      q16, k16, vT16, att16);

  gemm_wmma_kernel<2, true><<<gg, 128, 0, stream>>>(
      nullptr, att16, Wo, bo, nullptr, (float*)d_out, M_TOT, E_DIM, E_DIM);
}
